// GRU_55070070670225
// MI455X (gfx1250) — matrix operand order repaired and checked by reading, not compiled, not run on hardware
//
#include <hip/hip_runtime.h>
#include <hip/hip_bf16.h>
#include <math.h>

#define EMB   1536
#define HID   1536
#define VOCAB 32000
#define LAYERS 2
#define SEQ   64
#define BATCH 32

#define KCH   128            // K-chunk for TDM double buffering (bf16 elems)
#define NKCH  (HID / KCH)    // 12 chunks

typedef __attribute__((ext_vector_type(16))) __bf16 v16bf;
typedef __attribute__((ext_vector_type(8)))  __bf16 v8bf;
typedef __attribute__((ext_vector_type(8)))  float  v8f;

#if defined(__HIP_DEVICE_COMPILE__) && __has_builtin(__builtin_amdgcn_tensor_load_to_lds)
#define HAVE_TDM 1
#else
#define HAVE_TDM 0
#endif

static __device__ __forceinline__ v16bf cat8(v8bf lo, v8bf hi) {
  union { v16bf v; v8bf h[2]; } u;
  u.h[0] = lo; u.h[1] = hi;
  return u.v;
}

static __device__ __forceinline__ v8f wmma_bf16(v16bf a, v16bf b, v8f c) {
  // D = A(16x32 bf16) * B(32x16 bf16) + C(16x16 f32)
  return __builtin_amdgcn_wmma_f32_16x16x32_bf16(
      /*neg_a=*/false, a, /*neg_b=*/false, b,
      /*c_mod=*/(short)0, c, /*reuse_a=*/false, /*reuse_b=*/false);
}

#if HAVE_TDM
typedef unsigned int u32x4 __attribute__((ext_vector_type(4)));
typedef int          i32x4 __attribute__((ext_vector_type(4)));
typedef int          i32x8 __attribute__((ext_vector_type(8)));

// 2D tensor_load_to_lds: tile (tile_d0 x tile_d1), dims/strides in data_size
// units (code 3 = 8 bytes). D# per CDNA5 ISA.
// This toolchain exposes the 6-arg builtin (clang-23 / therock form):
//   (u32x4 group0, i32x8 group1, i32x4 group2, i32x4 group3, i32x8 pad, i32 cpol)
static __device__ __forceinline__ void tdm_load_2d(
    unsigned lds_addr, const void* gptr,
    unsigned tensor_d0, unsigned tensor_d1,
    unsigned stride_d0,
    unsigned tile_d0, unsigned tile_d1,
    unsigned data_size_code) {
  unsigned long long ga = (unsigned long long)(uintptr_t)gptr;
  u32x4 g0;
  g0[0] = 1u;                                   // count=1, is_restore=0, no gather
  g0[1] = lds_addr;                             // LDS byte address
  g0[2] = (unsigned)(ga & 0xffffffffu);         // global_addr[31:0]
  g0[3] = (unsigned)((ga >> 32) & 0x01ffffffu)  // global_addr[56:32]
          | (2u << 30);                         // type = 2 ("image")
  i32x8 g1;
  g1[0] = (int)(data_size_code << 16);          // wg_mask=0, no barrier/iter/pad
  g1[1] = (int)((tensor_d0 & 0xffffu) << 16);   // atomic_addr=0 | tensor_dim0[15:0]
  g1[2] = (int)((tensor_d0 >> 16) | ((tensor_d1 & 0xffffu) << 16));
  g1[3] = (int)((tensor_d1 >> 16) | (tile_d0 << 16));
  g1[4] = (int)(tile_d1 & 0xffffu);             // tile_dim2 = 0 (2D)
  g1[5] = (int)stride_d0;                       // tensor_dim0_stride[31:0]
  g1[6] = 0;                                    // stride hi | dim1_stride (unused, 2D)
  g1[7] = 0;
  i32x4 z4 = {0, 0, 0, 0};
  i32x8 z8 = {0, 0, 0, 0, 0, 0, 0, 0};
  __builtin_amdgcn_tensor_load_to_lds(g0, g1, z4, z4, z8, 0);
}

static __device__ __forceinline__ unsigned lds_of(const void* p) {
  return (unsigned)(uintptr_t)p;   // flat LDS address truncates to LDS offset
}
#endif

// ---------------------------------------------------------------------------
// Transpose + convert fp32 [K,N] row-major -> bf16 [N,K] row-major.
// ---------------------------------------------------------------------------
__global__ __launch_bounds__(256)
void transpose_to_bf16(const float* __restrict__ in, __bf16* __restrict__ out,
                       int K, int N) {
  __shared__ float tile[32][33];
  const int tx = threadIdx.x & 31, ty = threadIdx.x >> 5;
  const int nb = blockIdx.x * 32, kb = blockIdx.y * 32;
  #pragma unroll
  for (int r = ty; r < 32; r += 8)
    tile[r][tx] = in[(size_t)(kb + r) * N + (nb + tx)];
  __syncthreads();
  #pragma unroll
  for (int r = ty; r < 32; r += 8)
    out[(size_t)(nb + r) * K + (kb + tx)] = (__bf16)tile[tx][r];
}

// ---------------------------------------------------------------------------
// x0[s,b,:] = bf16( emb_table[tok[s,b]] * sqrt(EMB) )
// ---------------------------------------------------------------------------
__global__ __launch_bounds__(256)
void embed_scale_bf16(const int* __restrict__ tok, const float* __restrict__ emb,
                      __bf16* __restrict__ xout, float scale) {
  const int sb = blockIdx.x;
  const int t  = tok[sb];
  const float* src = emb + (size_t)t * EMB;
  __bf16* dst = xout + (size_t)sb * EMB;
  for (int i = threadIdx.x; i < EMB; i += blockDim.x)
    dst[i] = (__bf16)(src[i] * scale);
}

__global__ __launch_bounds__(256)
void init_hidden(const float* __restrict__ hidden, float* __restrict__ hstate,
                 __bf16* __restrict__ hb0, int n) {
  int i = blockIdx.x * blockDim.x + threadIdx.x;
  if (i < n) { float v = hidden[i]; hstate[i] = v; hb0[i] = (__bf16)v; }
}

// ---------------------------------------------------------------------------
// One GRU layer, one timestep. 8 waves; each wave owns 16 columns and runs
// all 6 gate GEMMs (two 16x16x32 bf16 WMMA tiles each, M=32). A-tiles (x,h)
// are DMA'd into LDS by the Tensor Data Mover, double-buffered on TENSORcnt,
// so the VMEM pipe is free for the B-fragment streams (L2-resident weights).
// ---------------------------------------------------------------------------
__global__ __launch_bounds__(256)
void gru_gate_kernel(const __bf16* __restrict__ xin,   // [B, HID]
                     const __bf16* __restrict__ hin,   // [B, HID]
                     float*        __restrict__ hstate,// [B, HID] f32 in-place
                     const __bf16* __restrict__ WT6,   // 6 x [HID, HID] bf16, N-major
                     const float* __restrict__ b_ir, const float* __restrict__ b_hr,
                     const float* __restrict__ b_iu, const float* __restrict__ b_hu,
                     const float* __restrict__ b_ic, const float* __restrict__ b_hc,
                     __bf16* __restrict__ hb_out,
                     __bf16* __restrict__ x_copy) {
  const int tid  = threadIdx.x;
  const int lane = tid & 31, wave = tid >> 5;
  const int ncol = blockIdx.x * 128 + wave * 16 + (lane & 15);
  const int kB   = (lane >> 4) * 16;
  const int klo  = (lane >> 4) * 8;
  const int am   = lane & 15;
  const size_t HH = (size_t)HID * HID;

  v8f zero = {};
  v8f acc[6][2];
  #pragma unroll
  for (int g = 0; g < 6; ++g) { acc[g][0] = zero; acc[g][1] = zero; }

#if HAVE_TDM
  __shared__ __attribute__((aligned(16))) __bf16 lX[2][BATCH * KCH];
  __shared__ __attribute__((aligned(16))) __bf16 lH[2][BATCH * KCH];
  const unsigned ROW8  = (HID * 2) / 8;   // row stride in 8B units (384)
  const unsigned TILE8 = (KCH * 2) / 8;   // chunk width in 8B units (32)

  if (wave == 0) {   // chunk 0 -> buffer 0
    tdm_load_2d(lds_of(&lX[0][0]), xin, TILE8, BATCH, ROW8, TILE8, BATCH, 3u);
    tdm_load_2d(lds_of(&lH[0][0]), hin, TILE8, BATCH, ROW8, TILE8, BATCH, 3u);
  }

  for (int c = 0; c < NKCH; ++c) {
    if (wave == 0) {
      if (c + 1 < NKCH) {   // prefetch next chunk into the other buffer
        int nb = (c + 1) & 1;
        tdm_load_2d(lds_of(&lX[nb][0]), xin + (size_t)(c + 1) * KCH,
                    TILE8, BATCH, ROW8, TILE8, BATCH, 3u);
        tdm_load_2d(lds_of(&lH[nb][0]), hin + (size_t)(c + 1) * KCH,
                    TILE8, BATCH, ROW8, TILE8, BATCH, 3u);
        __builtin_amdgcn_s_wait_tensorcnt(2);  // chunk c landed, c+1 in flight
      } else {
        __builtin_amdgcn_s_wait_tensorcnt(0);
      }
    }
    __syncthreads();

    const __bf16* lx = &lX[c & 1][0];
    const __bf16* lh = &lH[c & 1][0];
    #pragma unroll
    for (int ks = 0; ks < KCH; ks += 32) {
      v16bf ax[2], ah[2];
      #pragma unroll
      for (int mh = 0; mh < 2; ++mh) {
        int row = mh * 16 + am;
        ax[mh] = cat8(*(const v8bf*)&lx[row * KCH + ks + kB],
                      *(const v8bf*)&lx[row * KCH + ks + kB + 8]);
        ah[mh] = cat8(*(const v8bf*)&lh[row * KCH + ks + kB],
                      *(const v8bf*)&lh[row * KCH + ks + kB + 8]);
      }
      const int kb = c * KCH + ks;
      #pragma unroll
      for (int g = 0; g < 6; ++g) {
        const __bf16* wp = WT6 + (size_t)g * HH + (size_t)ncol * HID + kb + kB;
        v16bf bf = *(const v16bf*)wp;
        if (kb + 32 < HID) __builtin_prefetch(wp + 32, 0, 2);
        #pragma unroll
        for (int mh = 0; mh < 2; ++mh)
          acc[g][mh] = wmma_bf16((g & 1) ? ah[mh] : ax[mh], bf, acc[g][mh]);
      }
    }
    __syncthreads();   // buffer c&1 free for the issue at iter c+1
  }
#else
  // Fallback: per-32-slab VALU staging (known-good path)
  __shared__ __attribute__((aligned(16))) __bf16 lAx[BATCH * 32];
  __shared__ __attribute__((aligned(16))) __bf16 lAh[BATCH * 32];
  for (int kb = 0; kb < HID; kb += 32) {
    {
      int e = tid * 4;
      int m = e >> 5, kk = e & 31;
      *(uint2*)&lAx[e] = *(const uint2*)&xin[(size_t)m * HID + kb + kk];
      *(uint2*)&lAh[e] = *(const uint2*)&hin[(size_t)m * HID + kb + kk];
    }
    __syncthreads();
    v16bf ax[2], ah[2];
    #pragma unroll
    for (int mh = 0; mh < 2; ++mh) {
      int row = mh * 16 + am;
      ax[mh] = cat8(*(const v8bf*)&lAx[row * 32 + kB],
                    *(const v8bf*)&lAx[row * 32 + kB + 8]);
      ah[mh] = cat8(*(const v8bf*)&lAh[row * 32 + kB],
                    *(const v8bf*)&lAh[row * 32 + kB + 8]);
    }
    #pragma unroll
    for (int g = 0; g < 6; ++g) {
      const __bf16* wp = WT6 + (size_t)g * HH + (size_t)ncol * HID + kb + kB;
      v16bf bf = *(const v16bf*)wp;
      if (kb + 32 < HID) __builtin_prefetch(wp + 32, 0, 2);
      #pragma unroll
      for (int mh = 0; mh < 2; ++mh)
        acc[g][mh] = wmma_bf16((g & 1) ? ah[mh] : ax[mh], bf, acc[g][mh]);
    }
    __syncthreads();
  }
#endif

  // GRU nonlinearity, wave-local. C-layout: n = lane&15, m = vgpr + (lane>=16)*8.
  const float bir = b_ir[ncol], bhr = b_hr[ncol];
  const float biu = b_iu[ncol], bhu = b_hu[ncol];
  const float bic = b_ic[ncol], bhc = b_hc[ncol];
  const int mtop = (lane >> 4) * 8;
  #pragma unroll
  for (int mh = 0; mh < 2; ++mh) {
    #pragma unroll
    for (int i = 0; i < 8; ++i) {
      int m = mh * 16 + mtop + i;
      size_t idx = (size_t)m * HID + ncol;
      float hprev = hstate[idx];
      float r = 1.f / (1.f + __expf(-(acc[0][mh][i] + bir + acc[1][mh][i] + bhr)));
      float u = 1.f / (1.f + __expf(-(acc[2][mh][i] + biu + acc[3][mh][i] + bhu)));
      float c = tanhf(acc[4][mh][i] + bic + r * (acc[5][mh][i] + bhc));
      float hnew = (1.f - u) * c + u * hprev;
      hstate[idx] = hnew;
      __bf16 hb = (__bf16)hnew;
      hb_out[idx] = hb;
      if (x_copy) x_copy[idx] = hb;
    }
  }
}

// ---------------------------------------------------------------------------
// Batched logits GEMM: out[2048, 32000] = X @ W_out^T + b_out (W_out read once).
// ---------------------------------------------------------------------------
__global__ __launch_bounds__(256)
void gru_logits_kernel(const __bf16* __restrict__ X,   // [M, K] bf16
                       const __bf16* __restrict__ WT,  // [N, K] bf16
                       const float* __restrict__ bias, // [N]
                       float* __restrict__ out) {      // [M, N] f32
  const int K = HID, N = VOCAB;
  __shared__ __attribute__((aligned(16))) __bf16 lA[64 * 32];

  const int tid  = threadIdx.x;
  const int lane = tid & 31, wave = tid >> 5;
  const int wm = wave >> 2, wn = wave & 3;
  const int mblk  = blockIdx.y * 64;
  const int nwave = blockIdx.x * 128 + wn * 32;
  const int am  = lane & 15;
  const int klo = (lane >> 4) * 8;
  const int kB  = (lane >> 4) * 16;

  v8f zero = {};
  v8f acc[2][2] = {{zero, zero}, {zero, zero}};

  for (int kb = 0; kb < K; kb += 32) {
    {
      int e = tid * 8;
      int r = e >> 5, kk = e & 31;
      *(uint4*)&lA[e] = *(const uint4*)&X[(size_t)(mblk + r) * K + kb + kk];
    }
    __syncthreads();
    v16bf a[2];
    #pragma unroll
    for (int mh = 0; mh < 2; ++mh) {
      int row = wm * 32 + mh * 16 + am;
      a[mh] = cat8(*(const v8bf*)&lA[row * 32 + kB],
                   *(const v8bf*)&lA[row * 32 + kB + 8]);
    }
    #pragma unroll
    for (int nh = 0; nh < 2; ++nh) {
      int col = nwave + nh * 16 + am;
      const __bf16* wp = &WT[(size_t)col * K + kb + kB];
      v16bf b = *(const v16bf*)wp;
      if (kb + 32 < K) __builtin_prefetch(wp + 32, 0, 2);
      #pragma unroll
      for (int mh = 0; mh < 2; ++mh)
        acc[mh][nh] = wmma_bf16(a[mh], b, acc[mh][nh]);
    }
    __syncthreads();
  }

  const int mtop = (lane >> 4) * 8;
  #pragma unroll
  for (int nh = 0; nh < 2; ++nh) {
    int col = nwave + nh * 16 + am;
    float bo = bias[col];
    #pragma unroll
    for (int mh = 0; mh < 2; ++mh) {
      #pragma unroll
      for (int i = 0; i < 8; ++i) {
        int m = mblk + wm * 32 + mh * 16 + mtop + i;
        out[(size_t)m * N + col] = acc[mh][nh][i] + bo;
      }
    }
  }
}

__global__ __launch_bounds__(256)
void copy_hfinal(const float* __restrict__ h, float* __restrict__ out, int n) {
  int i = blockIdx.x * blockDim.x + threadIdx.x;
  if (i < n) out[i] = h[i];
}

// ---------------------------------------------------------------------------
// Host orchestration
// ---------------------------------------------------------------------------
extern "C" void kernel_launch(void* const* d_in, const int* in_sizes, int n_in,
                              void* d_out, int out_size, void* d_ws, size_t ws_size,
                              hipStream_t stream) {
  (void)in_sizes; (void)n_in; (void)out_size; (void)ws_size;
  const int*   tok    = (const int*)  d_in[0];
  const float* hidden = (const float*)d_in[1];
  const float* emb    = (const float*)d_in[2];
  const float* Wsrc[6] = { (const float*)d_in[3],  (const float*)d_in[5],
                           (const float*)d_in[7],  (const float*)d_in[9],
                           (const float*)d_in[11], (const float*)d_in[13] };
  const float* bsrc[6] = { (const float*)d_in[4],  (const float*)d_in[6],
                           (const float*)d_in[8],  (const float*)d_in[10],
                           (const float*)d_in[12], (const float*)d_in[14] };
  const float* W_out = (const float*)d_in[15];
  const float* b_out = (const float*)d_in[16];
  float* out = (float*)d_out;

  const size_t HH = (size_t)HID * HID;
  const size_t BH = (size_t)BATCH * HID;

  char* ws = (char*)d_ws;
  size_t off = 0;
  __bf16* Wrec = (__bf16*)(ws + off); off += (size_t)12 * HH * 2;          // 56.6 MB
  __bf16* Wout = (__bf16*)(ws + off); off += (size_t)VOCAB * HID * 2;      // 98.3 MB
  __bf16* Xemb = (__bf16*)(ws + off); off += (size_t)SEQ * BH * 2;         //  6.3 MB
  __bf16* Xall = (__bf16*)(ws + off); off += (size_t)SEQ * BH * 2;         //  6.3 MB
  float*  hst  = (float*) (ws + off); off += (size_t)LAYERS * BH * 4;      //  0.4 MB
  __bf16* hb   = (__bf16*)(ws + off); off += (size_t)2 * LAYERS * BH * 2;  //  0.4 MB
  // hb layout: [ping][layer][B*H]

  // 1) weights -> bf16, transposed to [N,K]
  for (int l = 0; l < LAYERS; ++l)
    for (int g = 0; g < 6; ++g)
      transpose_to_bf16<<<dim3(HID / 32, HID / 32), 256, 0, stream>>>(
          Wsrc[g] + (size_t)l * HH, Wrec + (size_t)(l * 6 + g) * HH, HID, HID);
  transpose_to_bf16<<<dim3(VOCAB / 32, HID / 32), 256, 0, stream>>>(
      W_out, Wout, HID, VOCAB);

  // 2) scaled embeddings for all timesteps
  embed_scale_bf16<<<SEQ * BATCH, 256, 0, stream>>>(tok, emb, Xemb,
                                                    sqrtf((float)EMB));

  // 3) hidden state init (fp32 + bf16 mirror in ping buffer 0)
  {
    int n = LAYERS * (int)BH;
    init_hidden<<<(n + 255) / 256, 256, 0, stream>>>(hidden, hst, hb, n);
  }

  // 4) sequential recurrence: 2 launches per timestep
  for (int s = 0; s < SEQ; ++s) {
    int p = s & 1, q = 1 - p;
    gru_gate_kernel<<<HID / 128, 256, 0, stream>>>(
        Xemb + (size_t)s * BH,
        hb + ((size_t)p * LAYERS + 0) * BH,
        hst,
        Wrec,
        bsrc[0], bsrc[1], bsrc[2], bsrc[3], bsrc[4], bsrc[5],
        hb + ((size_t)q * LAYERS + 0) * BH,
        (__bf16*)nullptr);
    gru_gate_kernel<<<HID / 128, 256, 0, stream>>>(
        hb + ((size_t)q * LAYERS + 0) * BH,
        hb + ((size_t)p * LAYERS + 1) * BH,
        hst + BH,
        Wrec + 6 * HH,
        bsrc[0] + HID, bsrc[1] + HID, bsrc[2] + HID,
        bsrc[3] + HID, bsrc[4] + HID, bsrc[5] + HID,
        hb + ((size_t)q * LAYERS + 1) * BH,
        Xall + (size_t)s * BH);
  }

  // 5) one big logits GEMM over all timesteps
  gru_logits_kernel<<<dim3(VOCAB / 128, (SEQ * BATCH) / 64), 256, 0, stream>>>(
      Xall, Wout, b_out, out);

  // 6) h_final appended after outputs
  {
    int n = LAYERS * (int)BH;
    copy_hfinal<<<(n + 255) / 256, 256, 0, stream>>>(
        hst, out + (size_t)SEQ * BATCH * VOCAB, n);
  }
}
